// BaseVectorQuantizer_73126113181951
// MI455X (gfx1250) — matrix operand order repaired and checked by reading, not compiled, not run on hardware
//
#include <hip/hip_runtime.h>

// MI455X / gfx1250, wave32. bf16 hi/lo-split WMMA VQ nearest-neighbor.
// v2: 2 row-tiles (32 rows) per wave -> halves B-operand fetch per WMMA.

typedef __attribute__((ext_vector_type(16))) __bf16    v16bf;
typedef __attribute__((ext_vector_type(8)))  float     v8f;
typedef __attribute__((ext_vector_type(8)))  unsigned  v8u;

__device__ __forceinline__ unsigned short f2bf(float f) {
  unsigned u = __float_as_uint(f);
  u += 0x7FFFu + ((u >> 16) & 1u);           // round-to-nearest-even
  return (unsigned short)(u >> 16);
}
__device__ __forceinline__ float bf2f(unsigned short b) {
  return __uint_as_float(((unsigned)b) << 16);
}

// ---------------------------------------------------------------------------
// Prep: e2[j] = ||codebook[j]||^2 (f32), and codebook split into bf16 hi/lo,
// stored pre-swizzled in the per-lane WMMA B-operand layout:
//   chunk index (kt*2+c), then lane (0..31), then 8 dwords (2 bf16 each).
// B 32x16 bf16 layout: col j = lane&15; lanes<16 hold K=0..7,16..23 of the chunk,
// lanes>=16 hold K=8..15,24..31; dword v<4 holds K pair (8h+2v, +1), v>=4 holds (16+8h+2(v-4), +1).
// ---------------------------------------------------------------------------
__global__ void vq_prep(const float* __restrict__ cb, float* __restrict__ e2,
                        unsigned* __restrict__ bhi, unsigned* __restrict__ blo) {
  int t = blockIdx.x * blockDim.x + threadIdx.x;
  if (t >= 1024 * 32) return;
  int j = t >> 5;          // codebook row 0..1023
  int p = t & 31;          // k-pair 0..31
  int k = p * 2;           // even k
  float f0 = cb[j * 64 + k], f1 = cb[j * 64 + k + 1];
  unsigned short h0 = f2bf(f0), h1 = f2bf(f1);
  unsigned short l0 = f2bf(f0 - bf2f(h0)), l1 = f2bf(f1 - bf2f(h1));
  int kt = j >> 4, jl = j & 15;
  int c = k >> 5, kc = k & 31, half = (kc >> 3) & 1, v = ((kc & 7) >> 1) + 4 * (kc >> 4);
  int lane = jl + 16 * half;
  size_t w = (((size_t)(kt * 2 + c)) * 32 + lane) * 8 + v;
  bhi[w] = (unsigned)h0 | ((unsigned)h1 << 16);
  blo[w] = (unsigned)l0 | ((unsigned)l1 << 16);
  if (p == 0) {
    float s = 0.f;
    #pragma unroll
    for (int d = 0; d < 64; ++d) { float cv = cb[j * 64 + d]; s += cv * cv; }
    e2[j] = s;
  }
}

// ---------------------------------------------------------------------------
// Main: one wave per 32 flattened rows (two 16-row WMMA A tiles).
// score[m][j] = e2[j] - 2 * x[m].e[j]   (||x||^2 dropped: row-constant)
// dot via split-bf16: x ~ hi+lo ; dot ~ hi*Bhi + lo*Bhi + hi*Blo
// ---------------------------------------------------------------------------
__global__ __launch_bounds__(256) void vq_main(
    const float* __restrict__ x, const float* __restrict__ cb,
    const float* __restrict__ e2, const unsigned* __restrict__ bhi,
    const unsigned* __restrict__ blo, float* __restrict__ out) {
  __shared__ unsigned sidx[256];                 // 8 waves * 32 rows
  const int lane = threadIdx.x & 31;
  const int wid  = threadIdx.x >> 5;
  const int gw   = blockIdx.x * 8 + wid;         // 32-row strip id (4096 total)
  const int n0 = gw << 5;                        // first flat row of strip
  const int b  = n0 >> 12;                       // H*W = 4096
  const int h  = (n0 >> 6) & 63;
  const int w0 = n0 & 63;                        // 32-aligned: 0 or 32
  const int m  = lane & 15;                      // A row within tile
  const int ko = (lane >> 4) << 3;               // A K sub-offset: 0 or 8

  // ---- load two 16x64 X tiles, split to bf16 hi/lo in WMMA A layout ----
  // A 16x32 bf16: lane m<16: v0..3 = K 0..7(+ko), v4..7 = K 16..23(+ko)
  v16bf Ahi[2][2], Alo[2][2];                    // [tile][chunk]
  #pragma unroll
  for (int t = 0; t < 2; ++t) {
    #pragma unroll
    for (int c = 0; c < 2; ++c) {
      v8u hu, lu;
      #pragma unroll
      for (int v = 0; v < 8; ++v) {
        int K = 32 * c + ((v < 4) ? (ko + 2 * v) : (16 + ko + 2 * (v - 4)));
        size_t i0 = ((size_t)(b * 64 + K) * 64 + h) * 64 + w0 + t * 16 + m;
        size_t i1 = ((size_t)(b * 64 + K + 1) * 64 + h) * 64 + w0 + t * 16 + m;
        float f0 = x[i0], f1 = x[i1];
        unsigned short h0 = f2bf(f0), h1 = f2bf(f1);
        unsigned short l0 = f2bf(f0 - bf2f(h0)), l1 = f2bf(f1 - bf2f(h1));
        hu[v] = (unsigned)h0 | ((unsigned)h1 << 16);
        lu[v] = (unsigned)l0 | ((unsigned)l1 << 16);
      }
      Ahi[t][c] = __builtin_bit_cast(v16bf, hu);
      Alo[t][c] = __builtin_bit_cast(v16bf, lu);
    }
  }

  // ---- running per-slot min over 64 codebook tiles ----
  float    minv[2][8];
  unsigned mini[2][8];
  #pragma unroll
  for (int t = 0; t < 2; ++t)
    #pragma unroll
    for (int r = 0; r < 8; ++r) { minv[t][r] = 3.4e38f; mini[t][r] = 0u; }

  for (int kt = 0; kt < 64; ++kt) {
    v8f acc0 = {}, acc1 = {};
    #pragma unroll
    for (int c = 0; c < 2; ++c) {
      size_t off = (((size_t)(kt * 2 + c)) * 32 + lane) * 8;
      v16bf Bh = __builtin_bit_cast(v16bf, *(const v8u*)(bhi + off));
      v16bf Bl = __builtin_bit_cast(v16bf, *(const v8u*)(blo + off));
      acc0 = __builtin_amdgcn_wmma_f32_16x16x32_bf16(
          false, Ahi[0][c], false, Bh, (short)0, acc0, false, false);
      acc1 = __builtin_amdgcn_wmma_f32_16x16x32_bf16(
          false, Ahi[1][c], false, Bh, (short)0, acc1, false, false);
      acc0 = __builtin_amdgcn_wmma_f32_16x16x32_bf16(
          false, Alo[0][c], false, Bh, (short)0, acc0, false, false);
      acc1 = __builtin_amdgcn_wmma_f32_16x16x32_bf16(
          false, Alo[1][c], false, Bh, (short)0, acc1, false, false);
      acc0 = __builtin_amdgcn_wmma_f32_16x16x32_bf16(
          false, Ahi[0][c], false, Bl, (short)0, acc0, false, false);
      acc1 = __builtin_amdgcn_wmma_f32_16x16x32_bf16(
          false, Ahi[1][c], false, Bl, (short)0, acc1, false, false);
    }
    unsigned jg  = (unsigned)(kt * 16 + m);
    float    e2v = e2[kt * 16 + m];
    #pragma unroll
    for (int r = 0; r < 8; ++r) {               // C/D slot r: row r(+8), col=lane&15
      float d0 = fmaf(-2.0f, acc0[r], e2v);
      float d1 = fmaf(-2.0f, acc1[r], e2v);
      bool l0 = d0 < minv[0][r];
      bool l1 = d1 < minv[1][r];
      minv[0][r] = l0 ? d0 : minv[0][r];
      mini[0][r] = l0 ? jg : mini[0][r];
      minv[1][r] = l1 ? d1 : minv[1][r];
      mini[1][r] = l1 ? jg : mini[1][r];
    }
  }

  // ---- butterfly min-reduce across the 16 j-lanes of each half ----
  #pragma unroll
  for (int t = 0; t < 2; ++t) {
    #pragma unroll
    for (int r = 0; r < 8; ++r) {
      #pragma unroll
      for (int off = 1; off <= 8; off <<= 1) {
        float    ov = __shfl_xor(minv[t][r], off, 32);
        unsigned oi = __shfl_xor(mini[t][r], off, 32);
        if (ov < minv[t][r] || (ov == minv[t][r] && oi < mini[t][r])) {
          minv[t][r] = ov; mini[t][r] = oi;
        }
      }
    }
  }

  // slot r @ lanes 0-15 -> row r ; @ lanes 16-31 -> row r+8. Bounce via LDS.
  #pragma unroll
  for (int t = 0; t < 2; ++t) {
    #pragma unroll
    for (int r = 0; r < 8; ++r) {
      if (lane == r)      sidx[wid * 32 + t * 16 + r]     = mini[t][r];
      if (lane == 16 + r) sidx[wid * 32 + t * 16 + 8 + r] = mini[t][r];
    }
  }
  __syncthreads();

  // ---- scatter winning codewords back out in BCHW ----
  const int dh = lane >> 4;                      // half-wave covers d parity
  #pragma unroll
  for (int t = 0; t < 2; ++t) {
    unsigned row = sidx[wid * 32 + t * 16 + m];
    const float* crow = cb + (size_t)row * 64;
    #pragma unroll
    for (int d0 = 0; d0 < 64; d0 += 2) {
      int dd = d0 + dh;
      out[((size_t)(b * 64 + dd) * 64 + h) * 64 + w0 + t * 16 + m] = crow[dd];
    }
  }
}

extern "C" void kernel_launch(void* const* d_in, const int* in_sizes, int n_in,
                              void* d_out, int out_size, void* d_ws, size_t ws_size,
                              hipStream_t stream) {
  (void)in_sizes; (void)n_in; (void)out_size; (void)ws_size;
  const float* x  = (const float*)d_in[0];   // [32,64,64,64] f32
  const float* cb = (const float*)d_in[1];   // [1024,64] f32
  float* out = (float*)d_out;

  // workspace: e2 (4KB) | Bhi (128KB) | Blo (128KB)
  float*    e2  = (float*)d_ws;
  unsigned* bhi = (unsigned*)((char*)d_ws + 4096);
  unsigned* blo = (unsigned*)((char*)d_ws + 4096 + 131072);

  vq_prep<<<128, 256, 0, stream>>>(cb, e2, bhi, blo);
  // 131072 rows / 32 per wave / 8 waves per block = 512 blocks
  vq_main<<<512, 256, 0, stream>>>(x, cb, e2, bhi, blo, out);
}
